// TransMILBlock_19104014532631
// MI455X (gfx1250) — compile-verified, matrix operand order checked by reading, not run on hardware
//
#include <hip/hip_runtime.h>
#include <hip/hip_bf16.h>

// ---------------- constants ----------------
#define B_   4
#define N_   9216
#define D_   512
#define NH_  8
#define HD_  64
#define L_   64
#define MLP_ 2048
#define HW_  96
#define MN_  (B_ * N_)   // 36864 rows

typedef __attribute__((ext_vector_type(16))) __bf16 v16bf;
typedef __attribute__((ext_vector_type(8)))  __bf16 v8bf;
typedef __attribute__((ext_vector_type(8)))  float  v8f;

// ---------------- helpers ----------------
static __device__ __forceinline__ v8f wmma_bf16(v16bf a, v16bf b, v8f c) {
    // (neg_a, A, neg_b, B, c_mod, C, reuse_a, reuse_b)
    return __builtin_amdgcn_wmma_f32_16x16x32_bf16(false, a, false, b, (short)0, c, false, false);
}

// Load a 16-lane-striped bf16 fragment from a row-major matrix.
// Per CDNA5 16-bit A/B layout: lanes 0-15 hold K={k0..k0+7, k0+16..k0+23},
// lanes 16-31 hold K={k0+8..k0+15, k0+24..k0+31} of row `row`.
static __device__ __forceinline__ v16bf load_frag(const __bf16* base, int ld, int row, int k0, int half) {
    const __bf16* p = base + (size_t)row * (size_t)ld + (size_t)(k0 + half * 8);
    v8bf lo = *(const v8bf*)p;
    v8bf hi = *(const v8bf*)(p + 16);
    v16bf f;
#pragma unroll
    for (int e = 0; e < 8; ++e) { f[e] = lo[e]; f[8 + e] = hi[e]; }
    return f;
}

// CDNA5 async copy global -> LDS (16B per lane), tracked by ASYNCcnt.
static __device__ __forceinline__ void async_ld16(const __bf16* g, __bf16* l) {
    unsigned lds = (unsigned)(uintptr_t)l;   // low 32 bits of flat addr == wave-relative LDS offset
    asm volatile("global_load_async_to_lds_b128 %0, %1, off" :: "v"(lds), "v"(g) : "memory");
}

static __device__ __forceinline__ float block_sum512(float v, float* red) {
    int t = threadIdx.x;
    red[t] = v; __syncthreads();
#pragma unroll
    for (int s = 256; s > 0; s >>= 1) {
        if (t < s) red[t] += red[t + s];
        __syncthreads();
    }
    float r = red[0]; __syncthreads();
    return r;
}

// ---------------- 1) fp32 -> bf16 weight convert ----------------
__global__ void f32_to_bf16_kernel(const float* __restrict__ src, __bf16* __restrict__ dst, int n) {
    int i = blockIdx.x * 256 + threadIdx.x;
    if (i < n) dst[i] = (__bf16)src[i];
}

// ---------------- 2) PPEG depthwise convs + residual + 2x LayerNorm ----------------
__global__ __launch_bounds__(512) void ppeg_ln_kernel(
    const float* __restrict__ x,
    const float* __restrict__ w3, const float* __restrict__ b3,
    const float* __restrict__ w5, const float* __restrict__ b5,
    const float* __restrict__ w7, const float* __restrict__ b7,
    const float* __restrict__ pg, const float* __restrict__ pb,
    const float* __restrict__ n1g, const float* __restrict__ n1b,
    float* __restrict__ x1, __bf16* __restrict__ xn)
{
    __shared__ float red[512];
    int b = blockIdx.x / N_;
    int n = blockIdx.x % N_;
    int hr = n / HW_, wc = n % HW_;
    int d = threadIdx.x;
    const float* xb = x + (size_t)b * N_ * D_;

    float pos = 0.f;
#pragma unroll
    for (int dy = -3; dy <= 3; ++dy) {
#pragma unroll
        for (int dx = -3; dx <= 3; ++dx) {
            int yy = hr + dy, xx = wc + dx;
            if (yy < 0 || yy >= HW_ || xx < 0 || xx >= HW_) continue;
            float xv = xb[((size_t)(yy * HW_ + xx)) * D_ + d];
            float wsum = w7[d * 49 + (dy + 3) * 7 + (dx + 3)];
            if (dy >= -2 && dy <= 2 && dx >= -2 && dx <= 2) wsum += w5[d * 25 + (dy + 2) * 5 + (dx + 2)];
            if (dy >= -1 && dy <= 1 && dx >= -1 && dx <= 1) wsum += w3[d * 9 + (dy + 1) * 3 + (dx + 1)];
            pos = fmaf(xv, wsum, pos);
        }
    }
    pos += b3[d] + b5[d] + b7[d];
    float t = xb[(size_t)n * D_ + d] + pos;

    // LN (ppeg)
    float s1 = block_sum512(t, red);
    float s2 = block_sum512(t * t, red);
    float mean = s1 * (1.f / D_);
    float var  = s2 * (1.f / D_) - mean * mean;
    float y = (t - mean) * rsqrtf(var + 1e-5f) * pg[d] + pb[d];
    x1[((size_t)b * N_ + n) * D_ + d] = y;

    // LN (norm1)
    s1 = block_sum512(y, red);
    s2 = block_sum512(y * y, red);
    mean = s1 * (1.f / D_);
    var  = s2 * (1.f / D_) - mean * mean;
    float z = (y - mean) * rsqrtf(var + 1e-5f) * n1g[d] + n1b[d];
    xn[((size_t)b * N_ + n) * D_ + d] = (__bf16)z;
}

// ---------------- standalone LayerNorm (norm2) ----------------
__global__ __launch_bounds__(512) void ln_kernel(
    const float* __restrict__ xin, const float* __restrict__ g, const float* __restrict__ bb,
    __bf16* __restrict__ out)
{
    __shared__ float red[512];
    size_t row = blockIdx.x;
    int d = threadIdx.x;
    float t = xin[row * D_ + d];
    float s1 = block_sum512(t, red);
    float s2 = block_sum512(t * t, red);
    float mean = s1 * (1.f / D_);
    float var  = s2 * (1.f / D_) - mean * mean;
    out[row * D_ + d] = (__bf16)((t - mean) * rsqrtf(var + 1e-5f) * g[d] + bb[d]);
}

// ---------------- 3) WMMA GEMM with double-buffered async-LDS staging ----------------
// C = act(A @ W^T + bias [+ res]); A:(M,K) bf16, W:(Nout,K) bf16, both row-major, lda=ldw=K.
// Block tile 128x64, 8 waves; per K-step: 8KB A-tile + 4KB B-tile staged via
// GLOBAL_LOAD_ASYNC_TO_LDS_B128 (ASYNCcnt), double buffered (24KB LDS).
#define GF_GELU 1
#define GF_RES  2
#define GF_F32  4
__global__ __launch_bounds__(256) void gemm_kernel(
    const __bf16* __restrict__ A, int K,
    const __bf16* __restrict__ W,
    const float* __restrict__ bias,
    const float* __restrict__ res,
    float* __restrict__ outF, __bf16* __restrict__ outB,
    int Nout, int flags)
{
    __shared__ __bf16 As[2][128 * 32];
    __shared__ __bf16 Bs[2][64 * 32];

    int tid = threadIdx.x;
    int w = tid >> 5, lane = tid & 31;
    int half = lane >> 4, lm = lane & 15;
    int mbase = blockIdx.y * 128;
    int nbase = blockIdx.x * 64;

    // cooperative async stage of one K-slab (3 x b128 per thread)
    auto issue = [&](int buf, int k0) {
        int c0 = tid;                    // A chunk 0..255
        int r0 = c0 >> 2, cc0 = c0 & 3;
        async_ld16(A + (size_t)(mbase + r0) * K + k0 + cc0 * 8, &As[buf][r0 * 32 + cc0 * 8]);
        int c1 = tid + 256;              // A chunk 256..511
        int r1 = c1 >> 2, cc1 = c1 & 3;
        async_ld16(A + (size_t)(mbase + r1) * K + k0 + cc1 * 8, &As[buf][r1 * 32 + cc1 * 8]);
        int r2 = tid >> 2, cc2 = tid & 3; // B chunk 0..255
        async_ld16(W + (size_t)(nbase + r2) * K + k0 + cc2 * 8, &Bs[buf][r2 * 32 + cc2 * 8]);
    };

    v8f acc[4] = {};
    const int nk = K >> 5;
    issue(0, 0);
    for (int ks = 0; ks < nk; ++ks) {
        int buf = ks & 1;
        if (ks + 1 < nk) {
            issue(buf ^ 1, (ks + 1) * 32);
            asm volatile("s_wait_asynccnt 0x3" ::: "memory");  // drain current buffer's 3 copies
        } else {
            asm volatile("s_wait_asynccnt 0x0" ::: "memory");
        }
        __syncthreads();                                       // LDS tile visible to all waves
        v16bf af = load_frag(&As[buf][0], 32, w * 16 + lm, 0, half);
#pragma unroll
        for (int t = 0; t < 4; ++t) {
            v16bf bfr = load_frag(&Bs[buf][0], 32, t * 16 + lm, 0, half);
            acc[t] = wmma_bf16(af, bfr, acc[t]);
        }
        __syncthreads();                                       // done reading before overwrite
    }

#pragma unroll
    for (int t = 0; t < 4; ++t) {
        int col = nbase + t * 16 + lm;
        float bv = bias[col];
#pragma unroll
        for (int r = 0; r < 8; ++r) {
            size_t row = (size_t)mbase + w * 16 + r + half * 8;
            float v = acc[t][r] + bv;
            if (flags & GF_RES) v += res[row * Nout + col];
            if (flags & GF_GELU) v = 0.5f * v * (1.f + erff(v * 0.70710678118f));
            if (flags & GF_F32) outF[row * Nout + col] = v;
            else                outB[row * Nout + col] = (__bf16)v;
        }
    }
}

// ---------------- 4) landmark gather (ql, kl) ----------------
__global__ void extract_landmarks_kernel(const __bf16* __restrict__ qkv, __bf16* __restrict__ land) {
    int i = blockIdx.x * 256 + threadIdx.x;
    if (i >= B_ * NH_ * L_ * HD_) return;
    int d = i & 63; int rest = i >> 6;
    int l = rest & 63; rest >>= 6;
    int h = rest & 7; int b = rest >> 3;
    int n = (int)((double)l * (double)(N_ - 1) / (double)(L_ - 1));
    size_t src = ((size_t)(b * N_ + n)) * (3 * D_) + h * HD_ + d;
    land[i]                                   = qkv[src];          // ql
    land[(size_t)B_ * NH_ * L_ * HD_ + i]     = qkv[src + D_];     // kl
}

// ---------------- 5) k1 = softmax(q @ kl^T * scale) ----------------
__global__ __launch_bounds__(256) void k1_kernel(
    const __bf16* __restrict__ qkv, const __bf16* __restrict__ kl, __bf16* __restrict__ k1)
{
    int bh = blockIdx.x, b = bh / NH_, h = bh % NH_;
    int w = threadIdx.x >> 5, lane = threadIdx.x & 31;
    int half = lane >> 4, lm = lane & 15;
    int nrow = blockIdx.y * 128 + w * 16;
    const __bf16* qb  = qkv + (size_t)b * N_ * (3 * D_) + h * HD_;
    const __bf16* klb = kl + (size_t)bh * L_ * HD_;

    v8f acc[4] = {};
#pragma unroll
    for (int k0 = 0; k0 < HD_; k0 += 32) {
        v16bf af = load_frag(qb, 3 * D_, nrow + lm, k0, half);
#pragma unroll
        for (int t = 0; t < 4; ++t) {
            v16bf bfr = load_frag(klb, HD_, t * 16 + lm, k0, half);
            acc[t] = wmma_bf16(af, bfr, acc[t]);
        }
    }
#pragma unroll
    for (int t = 0; t < 4; ++t)
#pragma unroll
        for (int r = 0; r < 8; ++r) acc[t][r] *= 0.125f;

#pragma unroll
    for (int r = 0; r < 8; ++r) {
        float mx = -1e30f;
#pragma unroll
        for (int t = 0; t < 4; ++t) mx = fmaxf(mx, acc[t][r]);
#pragma unroll
        for (int m = 1; m <= 8; m <<= 1) mx = fmaxf(mx, __shfl_xor(mx, m, 32));
        float e[4]; float s = 0.f;
#pragma unroll
        for (int t = 0; t < 4; ++t) { e[t] = expf(acc[t][r] - mx); s += e[t]; }
#pragma unroll
        for (int m = 1; m <= 8; m <<= 1) s += __shfl_xor(s, m, 32);
        float inv = 1.f / s;
        size_t row = (size_t)nrow + r + half * 8;
#pragma unroll
        for (int t = 0; t < 4; ++t)
            k1[((size_t)bh * N_ + row) * L_ + t * 16 + lm] = (__bf16)(e[t] * inv);
    }
}

// ---------------- 6) k2 softmax + Newton-Schulz pinv ----------------
__global__ __launch_bounds__(256) void k2pinv_kernel(
    const __bf16* __restrict__ ql, const __bf16* __restrict__ kl, float* __restrict__ k2inv)
{
    __shared__ float Am[64 * 64], Vm[64 * 64], Tm[64 * 64];
    __shared__ float rs[64], cs[64];
    int bh = blockIdx.x, tid = threadIdx.x;
    const __bf16* qlb = ql + (size_t)bh * 4096;
    const __bf16* klb = kl + (size_t)bh * 4096;

    for (int idx = tid; idx < 4096; idx += 256) {
        int i = idx >> 6, j = idx & 63;
        float s = 0.f;
        for (int d = 0; d < 64; ++d) s += (float)qlb[i * 64 + d] * (float)klb[j * 64 + d];
        Am[idx] = s * 0.125f;
    }
    __syncthreads();
    if (tid < 64) {
        float mx = -1e30f;
        for (int j = 0; j < 64; ++j) mx = fmaxf(mx, Am[tid * 64 + j]);
        float s = 0.f;
        for (int j = 0; j < 64; ++j) { float e = expf(Am[tid * 64 + j] - mx); Am[tid * 64 + j] = e; s += e; }
        float inv = 1.f / s;
        for (int j = 0; j < 64; ++j) Am[tid * 64 + j] *= inv;
        Am[tid * 64 + tid] += 1e-6f;  // + eps*I as in reference
    }
    __syncthreads();
    if (tid < 64) {
        float r = 0.f, c = 0.f;
        for (int j = 0; j < 64; ++j) { r += fabsf(Am[tid * 64 + j]); c += fabsf(Am[j * 64 + tid]); }
        rs[tid] = r; cs[tid] = c;
    }
    __syncthreads();
    if (tid == 0) {
        float mr = 0.f, mc = 0.f;
        for (int j = 0; j < 64; ++j) { mr = fmaxf(mr, rs[j]); mc = fmaxf(mc, cs[j]); }
        rs[0] = 1.f / (mr * mc);
    }
    __syncthreads();
    float alpha = rs[0];
    for (int idx = tid; idx < 4096; idx += 256)
        Vm[idx] = Am[(idx & 63) * 64 + (idx >> 6)] * alpha;  // V0 = A^T / (|A|_1 |A|_inf)
    __syncthreads();

    for (int it = 0; it < 12; ++it) {
        for (int idx = tid; idx < 4096; idx += 256) {      // T = A @ V
            int i = idx >> 6, j = idx & 63; float s = 0.f;
            for (int k = 0; k < 64; ++k) s += Am[i * 64 + k] * Vm[k * 64 + j];
            Tm[idx] = s;
        }
        __syncthreads();
        float pr[16];
#pragma unroll
        for (int q = 0; q < 16; ++q) {                     // P = V @ T (in regs)
            int idx = tid + q * 256; int i = idx >> 6, j = idx & 63; float s = 0.f;
            for (int k = 0; k < 64; ++k) s += Vm[i * 64 + k] * Tm[k * 64 + j];
            pr[q] = s;
        }
        __syncthreads();
#pragma unroll
        for (int q = 0; q < 16; ++q) Tm[tid + q * 256] = pr[q];
        __syncthreads();
        for (int idx = tid; idx < 4096; idx += 256)        // V = 2V - P
            Vm[idx] = 2.f * Vm[idx] - Tm[idx];
        __syncthreads();
    }
    for (int idx = tid; idx < 4096; idx += 256) k2inv[(size_t)bh * 4096 + idx] = Vm[idx];
}

// ---------------- 7) streaming kv = softmax(ql @ k^T) @ v  (flash-style) ----------------
__global__ __launch_bounds__(128) void k3v_kernel(
    const __bf16* __restrict__ qkv, const __bf16* __restrict__ ql, float* __restrict__ kv)
{
    __shared__ __bf16 pbuf[4][16 * 32];
    int bh = blockIdx.x, b = bh / NH_, h = bh % NH_;
    int w = threadIdx.x >> 5, lane = threadIdx.x & 31;
    int half = lane >> 4, lm = lane & 15;
    const __bf16* qlb = ql + (size_t)bh * L_ * HD_;
    const __bf16* kb  = qkv + (size_t)b * N_ * (3 * D_) + D_ + h * HD_;
    const __bf16* vb  = qkv + (size_t)b * N_ * (3 * D_) + 2 * D_ + h * HD_;

    v16bf aq0 = load_frag(qlb, HD_, w * 16 + lm, 0, half);
    v16bf aq1 = load_frag(qlb, HD_, w * 16 + lm, 32, half);

    v8f acc[4] = {};
    float rmax[8], rsum[8];
#pragma unroll
    for (int r = 0; r < 8; ++r) { rmax[r] = -1e30f; rsum[r] = 0.f; }

    for (int nc = 0; nc < N_; nc += 32) {
        v8f s[2] = {};
#pragma unroll
        for (int st = 0; st < 2; ++st) {
            v16bf b0 = load_frag(kb, 3 * D_, nc + st * 16 + lm, 0, half);
            s[st] = wmma_bf16(aq0, b0, s[st]);
            v16bf b1 = load_frag(kb, 3 * D_, nc + st * 16 + lm, 32, half);
            s[st] = wmma_bf16(aq1, b1, s[st]);
        }
        // online softmax per local row r (lanes 0-15: row r; 16-31: row r+8)
#pragma unroll
        for (int r = 0; r < 8; ++r) {
            float v0 = s[0][r] * 0.125f, v1 = s[1][r] * 0.125f;
            float mx = fmaxf(v0, v1);
#pragma unroll
            for (int m = 1; m <= 8; m <<= 1) mx = fmaxf(mx, __shfl_xor(mx, m, 32));
            float nm = fmaxf(rmax[r], mx);
            float f  = expf(rmax[r] - nm);
            float p0 = expf(v0 - nm), p1 = expf(v1 - nm);
            float ps = p0 + p1;
#pragma unroll
            for (int m = 1; m <= 8; m <<= 1) ps += __shfl_xor(ps, m, 32);
            rsum[r] = rsum[r] * f + ps;
            rmax[r] = nm;
#pragma unroll
            for (int t = 0; t < 4; ++t) acc[t][r] *= f;
            s[0][r] = p0; s[1][r] = p1;
        }
        // transpose P through per-wave LDS: C-layout -> A-fragment layout
#pragma unroll
        for (int st = 0; st < 2; ++st)
#pragma unroll
            for (int r = 0; r < 8; ++r)
                pbuf[w][(r + half * 8) * 32 + st * 16 + lm] = (__bf16)s[st][r];
        asm volatile("s_wait_dscnt 0x0" ::: "memory");  // per-wave LDS RAW: CDNA5 split counter
        v16bf pf;
#pragma unroll
        for (int e = 0; e < 16; ++e) {
            int kk = (e < 8) ? (half * 8 + e) : (16 + half * 8 + (e - 8));
            pf[e] = pbuf[w][lm * 32 + kk];
        }
        // acc += P @ V_chunk
#pragma unroll
        for (int t = 0; t < 4; ++t) {
            v16bf vf;
#pragma unroll
            for (int e = 0; e < 16; ++e) {
                int kk = (e < 8) ? (half * 8 + e) : (16 + half * 8 + (e - 8));
                vf[e] = vb[(size_t)(nc + kk) * (3 * D_) + t * 16 + lm];
            }
            acc[t] = wmma_bf16(pf, vf, acc[t]);
        }
    }
    float* out = kv + (size_t)bh * L_ * HD_;
#pragma unroll
    for (int r = 0; r < 8; ++r) {
        float inv = 1.f / rsum[r];
        int row = w * 16 + r + half * 8;
#pragma unroll
        for (int t = 0; t < 4; ++t)
            out[row * HD_ + t * 16 + lm] = acc[t][r] * inv;
    }
}

// ---------------- 8) M = k2inv @ kv, stored transposed (HD x L) in bf16 ----------------
__global__ __launch_bounds__(256) void mt_kernel(
    const float* __restrict__ k2inv, const float* __restrict__ kv, __bf16* __restrict__ Mt)
{
    int bh = blockIdx.x, tid = threadIdx.x;
    const float* Vi = k2inv + (size_t)bh * 4096;
    const float* Kv = kv + (size_t)bh * 4096;
    for (int idx = tid; idx < 4096; idx += 256) {
        int l = idx >> 6, d = idx & 63;
        float s = 0.f;
        for (int j = 0; j < 64; ++j) s += Vi[l * 64 + j] * Kv[j * 64 + d];
        Mt[(size_t)bh * 4096 + d * 64 + l] = (__bf16)s;
    }
}

// ---------------- 9) attnout = k1 @ M (scatter into head-interleaved layout) ----------------
__global__ __launch_bounds__(256) void attnout_kernel(
    const __bf16* __restrict__ k1, const __bf16* __restrict__ Mt, __bf16* __restrict__ attnout)
{
    int bh = blockIdx.x, b = bh / NH_, h = bh % NH_;
    int w = threadIdx.x >> 5, lane = threadIdx.x & 31;
    int half = lane >> 4, lm = lane & 15;
    int nrow = blockIdx.y * 128 + w * 16;
    const __bf16* k1b = k1 + (size_t)bh * N_ * L_;
    const __bf16* mtb = Mt + (size_t)bh * 4096;

    v8f acc[4] = {};
#pragma unroll
    for (int k0 = 0; k0 < L_; k0 += 32) {
        v16bf af = load_frag(k1b, L_, nrow + lm, k0, half);
#pragma unroll
        for (int t = 0; t < 4; ++t) {
            v16bf bfr = load_frag(mtb, L_, t * 16 + lm, k0, half);
            acc[t] = wmma_bf16(af, bfr, acc[t]);
        }
    }
#pragma unroll
    for (int r = 0; r < 8; ++r) {
        size_t row = (size_t)nrow + r + half * 8;
#pragma unroll
        for (int t = 0; t < 4; ++t)
            attnout[((size_t)b * N_ + row) * D_ + h * HD_ + t * 16 + lm] = (__bf16)acc[t][r];
    }
}

// ---------------- host launcher ----------------
extern "C" void kernel_launch(void* const* d_in, const int* in_sizes, int n_in,
                              void* d_out, int out_size, void* d_ws, size_t ws_size,
                              hipStream_t stream) {
    const float* x       = (const float*)d_in[0];
    const float* conv1_w = (const float*)d_in[1];
    const float* conv1_b = (const float*)d_in[2];
    const float* conv2_w = (const float*)d_in[3];
    const float* conv2_b = (const float*)d_in[4];
    const float* conv3_w = (const float*)d_in[5];
    const float* conv3_b = (const float*)d_in[6];
    const float* ppeg_g  = (const float*)d_in[7];
    const float* ppeg_b  = (const float*)d_in[8];
    const float* norm1_g = (const float*)d_in[9];
    const float* norm1_b = (const float*)d_in[10];
    const float* norm2_g = (const float*)d_in[11];
    const float* norm2_b = (const float*)d_in[12];
    const float* qkv_w   = (const float*)d_in[13];
    const float* qkv_b   = (const float*)d_in[14];
    const float* proj_w  = (const float*)d_in[15];
    const float* proj_b  = (const float*)d_in[16];
    const float* mlp_w1  = (const float*)d_in[17];
    const float* mlp_b1  = (const float*)d_in[18];
    const float* mlp_w2  = (const float*)d_in[19];
    const float* mlp_b2  = (const float*)d_in[20];

    char* base = (char*)d_ws;
    size_t off = 0;
    auto alloc = [&](size_t bytes) { size_t o = off; off = (off + bytes + 255) & ~(size_t)255; return o; };

    size_t o_x1   = alloc((size_t)MN_ * D_ * 4);          // residual stream (fp32), reused as x2
    size_t o_xn   = alloc((size_t)MN_ * D_ * 2);          // LN output bf16, reused for xn2
    size_t o_qkv  = alloc((size_t)MN_ * 3 * D_ * 2);      // qkv bf16
    size_t o_k1   = alloc((size_t)B_ * NH_ * N_ * L_ * 2);
    size_t o_land = alloc((size_t)2 * B_ * NH_ * L_ * HD_ * 2);
    size_t o_kv   = alloc((size_t)B_ * NH_ * L_ * HD_ * 4);
    size_t o_k2i  = alloc((size_t)B_ * NH_ * L_ * L_ * 4);
    size_t o_Mt   = alloc((size_t)B_ * NH_ * HD_ * L_ * 2);
    size_t o_ao   = alloc((size_t)MN_ * D_ * 2);          // attention out bf16
    size_t o_h    = alloc((size_t)MN_ * MLP_ * 2);        // MLP hidden bf16
    size_t o_wq   = alloc((size_t)3 * D_ * D_ * 2);
    size_t o_wp   = alloc((size_t)D_ * D_ * 2);
    size_t o_w1   = alloc((size_t)MLP_ * D_ * 2);
    size_t o_w2   = alloc((size_t)D_ * MLP_ * 2);

    float*  x1  = (float*)(base + o_x1);
    __bf16* xn  = (__bf16*)(base + o_xn);
    __bf16* qkv = (__bf16*)(base + o_qkv);
    __bf16* k1  = (__bf16*)(base + o_k1);
    __bf16* ql  = (__bf16*)(base + o_land);
    __bf16* kl  = ql + (size_t)B_ * NH_ * L_ * HD_;
    float*  kv  = (float*)(base + o_kv);
    float*  k2i = (float*)(base + o_k2i);
    __bf16* Mt  = (__bf16*)(base + o_Mt);
    __bf16* ao  = (__bf16*)(base + o_ao);
    __bf16* hb  = (__bf16*)(base + o_h);
    __bf16* wqb = (__bf16*)(base + o_wq);
    __bf16* wpb = (__bf16*)(base + o_wp);
    __bf16* w1b = (__bf16*)(base + o_w1);
    __bf16* w2b = (__bf16*)(base + o_w2);

    // weight conversion fp32 -> bf16
    f32_to_bf16_kernel<<<(3 * D_ * D_ + 255) / 256, 256, 0, stream>>>(qkv_w, wqb, 3 * D_ * D_);
    f32_to_bf16_kernel<<<(D_ * D_ + 255) / 256, 256, 0, stream>>>(proj_w, wpb, D_ * D_);
    f32_to_bf16_kernel<<<(MLP_ * D_ + 255) / 256, 256, 0, stream>>>(mlp_w1, w1b, MLP_ * D_);
    f32_to_bf16_kernel<<<(D_ * MLP_ + 255) / 256, 256, 0, stream>>>(mlp_w2, w2b, D_ * MLP_);

    // PPEG + residual + LN(ppeg) + LN(norm1)
    ppeg_ln_kernel<<<MN_, 512, 0, stream>>>(x, conv1_w, conv1_b, conv2_w, conv2_b, conv3_w, conv3_b,
                                            ppeg_g, ppeg_b, norm1_g, norm1_b, x1, xn);

    // qkv = xn @ qkv_w^T + qkv_b
    gemm_kernel<<<dim3((3 * D_) / 64, MN_ / 128), 256, 0, stream>>>(
        xn, D_, wqb, qkv_b, nullptr, nullptr, qkv, 3 * D_, 0);

    // landmarks
    extract_landmarks_kernel<<<(B_ * NH_ * L_ * HD_ + 255) / 256, 256, 0, stream>>>(qkv, ql);

    // k1 = softmax(q @ kl^T)
    k1_kernel<<<dim3(B_ * NH_, N_ / 128), 256, 0, stream>>>(qkv, kl, k1);

    // k2 -> pinv (Newton-Schulz)
    k2pinv_kernel<<<B_ * NH_, 256, 0, stream>>>(ql, kl, k2i);

    // kv = softmax(ql @ k^T) @ v   (streaming)
    k3v_kernel<<<B_ * NH_, 128, 0, stream>>>(qkv, ql, kv);

    // M = k2inv @ kv (store transposed, bf16)
    mt_kernel<<<B_ * NH_, 256, 0, stream>>>(k2i, kv, Mt);

    // attnout = k1 @ M
    attnout_kernel<<<dim3(B_ * NH_, N_ / 128), 256, 0, stream>>>(k1, Mt, ao);

    // x2 = x1 + attnout @ proj_w^T + proj_b   (in-place into x1)
    gemm_kernel<<<dim3(D_ / 64, MN_ / 128), 256, 0, stream>>>(
        ao, D_, wpb, proj_b, x1, x1, nullptr, D_, GF_RES | GF_F32);

    // xn2 = LN(x2, norm2)
    ln_kernel<<<MN_, 512, 0, stream>>>(x1, norm2_g, norm2_b, xn);

    // h = gelu(xn2 @ mlp_w1^T + mlp_b1)
    gemm_kernel<<<dim3(MLP_ / 64, MN_ / 128), 256, 0, stream>>>(
        xn, D_, w1b, mlp_b1, nullptr, nullptr, hb, MLP_, GF_GELU);

    // out = x2 + h @ mlp_w2^T + mlp_b2
    gemm_kernel<<<dim3(D_ / 64, MN_ / 128), 256, 0, stream>>>(
        hb, MLP_, w2b, mlp_b2, x1, (float*)d_out, nullptr, D_, GF_RES | GF_F32);

    (void)in_sizes; (void)n_in; (void)out_size; (void)ws_size;
}
